// HRMChess_17824114279066
// MI455X (gfx1250) — matrix operand order repaired and checked by reading, not compiled, not run on hardware
//
#include <hip/hip_runtime.h>

// ---------------------------------------------------------------------------
// CDNA5 (gfx1250) bf16 WMMA implementation of the HRM-chess forward pass.
// wave32; WMMA shape 16x16x32 bf16 -> f32.  H=192 = 12 N-tiles x 6 K-chunks.
// ---------------------------------------------------------------------------

typedef __attribute__((ext_vector_type(16))) __bf16 bf16x16;
typedef __attribute__((ext_vector_type(8)))  __bf16 bf16x8;
typedef __attribute__((ext_vector_type(8)))  float  f32x8;

#define WMMA_BF16(a, b, c) \
  __builtin_amdgcn_wmma_f32_16x16x32_bf16(false, (a), false, (b), (short)0, (c), false, false)

__device__ __forceinline__ bf16x16 cat16(bf16x8 lo, bf16x8 hi) {
  return __builtin_shufflevector(lo, hi, 0, 1, 2, 3, 4, 5, 6, 7, 8, 9, 10, 11, 12, 13, 14, 15);
}

// A-fragment: 16xK row-major activation tile (LDS or global), row stride in elems.
// ISA layout: lane<16 holds K {0..7,16..23}, lane>=16 holds K {8..15,24..31} of row lane&15.
__device__ __forceinline__ bf16x16 load_A(const __bf16* base, int stride, int kbase) {
  const int lane = threadIdx.x & 31;
  const int m = lane & 15, half = lane >> 4;
  const __bf16* p = base + m * stride + kbase + half * 8;
  bf16x8 lo = *(const bf16x8*)(p);
  bf16x8 hi = *(const bf16x8*)(p + 16);
  return cat16(lo, hi);
}

// B-fragment from pre-transposed weight Wt (N rows x K cols, row-major, global).
// ISA layout: lane<16 holds K {0..7,16..23}, lane>=16 holds K {8..15,24..31} of column N=lane&15.
__device__ __forceinline__ bf16x16 load_B(const __bf16* Wt, int nbase, int Ks, int kbase) {
  const int lane = threadIdx.x & 31;
  const __bf16* p = Wt + (size_t)(nbase + (lane & 15)) * Ks + kbase + (lane >> 4) * 8;
  bf16x8 lo = *(const bf16x8*)(p);
  bf16x8 hi = *(const bf16x8*)(p + 16);
  return cat16(lo, hi);
}

// C tile (16x16 f32): VGPR r, lane L -> M = r + 8*(L>>4), N = L&15.
__device__ __forceinline__ void store_relu(__bf16* base, int stride, int nbase, const f32x8& c) {
  const int lane = threadIdx.x & 31;
  const int half = lane >> 4, n = lane & 15;
#pragma unroll
  for (int r = 0; r < 8; ++r)
    base[(r + 8 * half) * stride + nbase + n] = (__bf16)fmaxf(c[r], 0.0f);
}

__device__ __forceinline__ f32x8 bias_init(const float* b, int nbase) {
  float v = b[nbase + (threadIdx.x & 15)];
  f32x8 c;
#pragma unroll
  for (int r = 0; r < 8; ++r) c[r] = v;
  return c;
}

// acc[12] += act(16x192 slice starting at K=0 of act buffer) @ Wt[:, kw0:kw0+192]
__device__ __forceinline__ void gemm_acc(f32x8 (&acc)[12], const __bf16* act, int SA,
                                         const __bf16* Wt, int Ks, int kw0) {
  for (int kc = 0; kc < 6; ++kc) {
    bf16x16 a = load_A(act, SA, kc * 32);
#pragma unroll
    for (int n = 0; n < 12; ++n) {
      bf16x16 b = load_B(Wt, n * 16, Ks, kw0 + kc * 32);
      acc[n] = WMMA_BF16(a, b, acc[n]);
    }
  }
}

// out = relu(act(16x192, stride 200) @ Wt(192x192) + bias), stored bf16 stride 200
__device__ __forceinline__ void dense192_relu(const __bf16* act, __bf16* outb,
                                              const __bf16* Wt, const float* bias) {
  f32x8 acc[12];
#pragma unroll
  for (int n = 0; n < 12; ++n) acc[n] = bias_init(bias, n * 16);
  gemm_acc(acc, act, 200, Wt, 192, 0);
  __builtin_amdgcn_wave_barrier();
#pragma unroll
  for (int n = 0; n < 12; ++n) store_relu(outb, 200, n * 16, acc[n]);
  __builtin_amdgcn_wave_barrier();
}

// ---------------------------------------------------------------------------
// Prep kernels: weight transpose + bf16 convert; BN-folded conv as dense 64x64.
// ---------------------------------------------------------------------------
__global__ void k_transpose_bf16(__bf16* dst, const float* src, int In, int Out, int Kpad) {
  long i = (long)blockIdx.x * blockDim.x + threadIdx.x;
  long total = (long)Out * Kpad;
  if (i >= total) return;
  int o = (int)(i / Kpad), k = (int)(i % Kpad);
  dst[i] = (k < In) ? (__bf16)src[(size_t)k * Out + o] : (__bf16)0.0f;
}

__global__ void k_conv_fold(__bf16* Mt, float* effb, const float* conv_w, const float* conv_b,
                            const float* bn_s, const float* bn_b, const float* bn_m,
                            const float* bn_v) {
  int i = blockIdx.x * blockDim.x + threadIdx.x;  // 12288*64 elements
  if (i >= 12288 * 64) return;
  int q = i & 63;
  int row = i >> 6;            // c*64 + p
  int p = row & 63, c = row >> 6;
  int pr = p >> 3, pc = p & 7, qr = q >> 3, qc = q & 7;
  int dy = qr - pr, dx = qc - pc;
  float s = bn_s[c] * rsqrtf(bn_v[c] + 1e-5f);
  float val = 0.0f;
  if (dy >= -1 && dy <= 1 && dx >= -1 && dx <= 1)
    val = s * conv_w[c * 9 + (dy + 1) * 3 + (dx + 1)];
  Mt[i] = (__bf16)val;
  if (i < 192) effb[i] = (conv_b[i] - bn_m[i]) * (bn_s[i] * rsqrtf(bn_v[i] + 1e-5f)) + bn_b[i];
}

// ---------------------------------------------------------------------------
// Encoder: conv(as WMMA) -> fused fc1 (K=12352) -> fc2 -> be1 -> be2 -> lf
// 4 waves / WG, 16 rows / wave, 128 WGs.  Dynamic LDS = 4 * 11776 B.
// ---------------------------------------------------------------------------
__global__ __launch_bounds__(128) void k_encode(
    const float* x, const float* ep_w1, const float* ep_b1, const float* ep_w2,
    const float* ep_b2, const __bf16* Mt, const float* effb, const __bf16* fc1t,
    const float* fc_b1, const __bf16* fc2t, const float* fc_b2, const __bf16* be1t,
    const float* be_b1, const __bf16* be2t, const float* be_b2, const __bf16* lw1ct,
    const float* l_b1, float* lf) {
  extern __shared__ char smem[];
  const int wave = threadIdx.x >> 5, lane = threadIdx.x & 31;
  char* ws = smem + wave * 11776;
  __bf16* slab  = (__bf16*)ws;                  // 16 x 72 (board / ef / conv tile)
  __bf16* fslab = slab + 16 * 72;               // 16 x 200 activation scratch
  float*  t1f   = (float*)(ws + 2304 + 6400);   // 16 x 48
  const int rowbase = (blockIdx.x * 4 + wave) * 16;

  // ---- extra-info MLP (VALU; tiny) into slab (64-wide zero-padded) ----
  for (int i = lane; i < 16 * 48; i += 32) {
    int r = i / 48, j = i % 48;
    const float* xr = x + (size_t)(rowbase + r) * 72 + 64;
    float a = ep_b1[j];
#pragma unroll
    for (int k = 0; k < 8; ++k) a += xr[k] * ep_w1[k * 48 + j];
    t1f[i] = fmaxf(a, 0.0f);
  }
  __builtin_amdgcn_wave_barrier();
  for (int i = lane; i < 16 * 64; i += 32) {
    int r = i >> 6, j = i & 63;
    float a = 0.0f;
    if (j < 48) {
      a = ep_b2[j];
      for (int k = 0; k < 48; ++k) a += t1f[r * 48 + k] * ep_w2[k * 48 + j];
    }
    slab[r * 72 + j] = (__bf16)a;
  }
  __builtin_amdgcn_wave_barrier();

  // ---- fc1 accumulators (init = fc_b1), ef-tail contribution first ----
  f32x8 f1[12];
#pragma unroll
  for (int n = 0; n < 12; ++n) f1[n] = bias_init(fc_b1, n * 16);
  {
    bf16x16 a0 = load_A(slab, 72, 0);
    bf16x16 a1 = load_A(slab, 72, 32);
#pragma unroll
    for (int n = 0; n < 12; ++n) {
      f1[n] = WMMA_BF16(a0, load_B(fc1t, n * 16, 12352, 12288), f1[n]);
      f1[n] = WMMA_BF16(a1, load_B(fc1t, n * 16, 12352, 12320), f1[n]);
    }
  }
  __builtin_amdgcn_wave_barrier();

  // ---- board -> bf16 slab -> A fragments kept in VGPRs ----
  for (int i = lane; i < 16 * 64; i += 32) {
    int r = i >> 6, q = i & 63;
    slab[r * 72 + q] = (__bf16)x[(size_t)(rowbase + r) * 72 + q];
  }
  __builtin_amdgcn_wave_barrier();
  bf16x16 aB0 = load_A(slab, 72, 0);
  bf16x16 aB1 = load_A(slab, 72, 32);
  __builtin_amdgcn_wave_barrier();

  // ---- per-channel: conv tile via WMMA (BN folded into Mt/effb), then fc1 K-slice ----
  for (int c = 0; c < 192; ++c) {
    float eb = effb[c];
    f32x8 ct[4];
#pragma unroll
    for (int t = 0; t < 4; ++t) {
#pragma unroll
      for (int r = 0; r < 8; ++r) ct[t][r] = eb;
      ct[t] = WMMA_BF16(aB0, load_B(Mt, c * 64 + t * 16, 64, 0), ct[t]);
      ct[t] = WMMA_BF16(aB1, load_B(Mt, c * 64 + t * 16, 64, 32), ct[t]);
    }
    __builtin_amdgcn_wave_barrier();
#pragma unroll
    for (int t = 0; t < 4; ++t) store_relu(slab, 72, t * 16, ct[t]);
    __builtin_amdgcn_wave_barrier();
    bf16x16 a0 = load_A(slab, 72, 0);
    bf16x16 a1 = load_A(slab, 72, 32);
#pragma unroll
    for (int n = 0; n < 12; ++n) {
      f1[n] = WMMA_BF16(a0, load_B(fc1t, n * 16, 12352, c * 64), f1[n]);
      f1[n] = WMMA_BF16(a1, load_B(fc1t, n * 16, 12352, c * 64 + 32), f1[n]);
    }
    __builtin_amdgcn_wave_barrier();
  }

  // ---- relu(fc1) -> fc2 -> be1 -> be2 (in-place 16x200 slab chain) ----
#pragma unroll
  for (int n = 0; n < 12; ++n) store_relu(fslab, 200, n * 16, f1[n]);
  __builtin_amdgcn_wave_barrier();
  dense192_relu(fslab, fslab, fc2t, fc_b2);
  dense192_relu(fslab, fslab, be1t, be_b1);
  dense192_relu(fslab, fslab, be2t, be_b2);

  // ---- lf = f @ l_w1[384:576] + l_b1  (f32, no relu) ----
  f32x8 acc[12];
#pragma unroll
  for (int n = 0; n < 12; ++n) acc[n] = bias_init(l_b1, n * 16);
  gemm_acc(acc, fslab, 200, lw1ct, 192, 0);
  const int half = lane >> 4, nn = lane & 15;
#pragma unroll
  for (int n = 0; n < 12; ++n)
#pragma unroll
    for (int r = 0; r < 8; ++r)
      lf[(size_t)(rowbase + r + 8 * half) * 192 + n * 16 + nn] = acc[n][r];
}

// ---------------------------------------------------------------------------
// Recurrent core: 8 blocks of [8 L-steps + 1 H-step] == 64 L + 8 H steps.
// 2 waves / WG (LDS 62976 B), 16 rows / wave, 256 WGs. Weights stream from L2.
// ---------------------------------------------------------------------------
__global__ __launch_bounds__(64) void k_recur(const float* lf, const __bf16* lW1t,
                                              const __bf16* lW2t, const float* l_b2,
                                              const __bf16* hW1t, const __bf16* hW2t,
                                              const float* h_b1, const float* h_b2,
                                              __bf16* zHout) {
  extern __shared__ char smem[];
  const int wave = threadIdx.x >> 5, lane = threadIdx.x & 31;
  char* ws = smem + wave * 31488;
  __bf16* zLs = (__bf16*)ws;            // 16 x 200
  __bf16* zHs = zLs + 16 * 200;         // 16 x 200
  __bf16* hbs = zHs + 16 * 200;         // 16 x 200
  float*  lfs = (float*)(ws + 19200);   // 16 x 192
  const int rowbase = (blockIdx.x * 2 + wave) * 16;
  const int half = lane >> 4, nn = lane & 15;

  for (int i = lane; i < 16 * 200; i += 32) { zLs[i] = (__bf16)0.0f; zHs[i] = (__bf16)0.0f; }
  for (int i = lane; i < 16 * 192; i += 32)
    lfs[i] = lf[(size_t)(rowbase + i / 192) * 192 + (i % 192)];
  __builtin_amdgcn_wave_barrier();

  for (int blk = 0; blk < 8; ++blk) {
    for (int t = 0; t < 8; ++t) {
      // ---- L-step: h = relu(zL@W1a + zH@W1b + lf); zL = relu(h@W2 + b2) ----
      f32x8 acc[12];
#pragma unroll
      for (int n = 0; n < 12; ++n)
#pragma unroll
        for (int r = 0; r < 8; ++r)
          acc[n][r] = lfs[(r + 8 * half) * 192 + n * 16 + nn];
      gemm_acc(acc, zLs, 200, lW1t, 384, 0);
      gemm_acc(acc, zHs, 200, lW1t, 384, 192);
      __builtin_amdgcn_wave_barrier();
#pragma unroll
      for (int n = 0; n < 12; ++n) store_relu(hbs, 200, n * 16, acc[n]);
      __builtin_amdgcn_wave_barrier();
      dense192_relu(hbs, zLs, lW2t, l_b2);
    }
    // ---- H-step: h = relu(zH@W1a + zL@W1b + b1); zH = relu(h@W2 + b2) ----
    f32x8 acc[12];
#pragma unroll
    for (int n = 0; n < 12; ++n) acc[n] = bias_init(h_b1, n * 16);
    gemm_acc(acc, zHs, 200, hW1t, 384, 0);
    gemm_acc(acc, zLs, 200, hW1t, 384, 192);
    __builtin_amdgcn_wave_barrier();
#pragma unroll
    for (int n = 0; n < 12; ++n) store_relu(hbs, 200, n * 16, acc[n]);
    __builtin_amdgcn_wave_barrier();
    dense192_relu(hbs, zHs, hW2t, h_b2);
  }

  for (int i = lane; i < 16 * 192; i += 32)
    zHout[(size_t)(rowbase + i / 192) * 192 + (i % 192)] = zHs[(i / 192) * 200 + (i % 192)];
}

// ---------------------------------------------------------------------------
// Policy head: zH(16x192) @ polt(4096x192) + pol_b -> f32 logits (B,4096)
// ---------------------------------------------------------------------------
__global__ __launch_bounds__(256) void k_policy(const __bf16* zH, const __bf16* polt,
                                                const float* pol_b, float* out) {
  const int wave = threadIdx.x >> 5, lane = threadIdx.x & 31;
  const int rowbase = (blockIdx.x * 8 + wave) * 16;
  const __bf16* zrow = zH + (size_t)rowbase * 192;
  bf16x16 a[6];
#pragma unroll
  for (int kc = 0; kc < 6; ++kc) a[kc] = load_A(zrow, 192, kc * 32);
  const int half = lane >> 4, nn = lane & 15;
  for (int n = 0; n < 256; ++n) {
    f32x8 acc = bias_init(pol_b, n * 16);
#pragma unroll
    for (int kc = 0; kc < 6; ++kc)
      acc = WMMA_BF16(a[kc], load_B(polt, n * 16, 192, kc * 32), acc);
#pragma unroll
    for (int r = 0; r < 8; ++r)
      out[(size_t)(rowbase + r + 8 * half) * 4096 + n * 16 + nn] = acc[r];
  }
}

// ---------------------------------------------------------------------------
// Value head (tiny): scalar per-row MLP 192->96->48->1, tanh.
// ---------------------------------------------------------------------------
__global__ __launch_bounds__(256) void k_value(const __bf16* zH, const float* w1,
                                               const float* b1, const float* w2,
                                               const float* b2, const float* w3,
                                               const float* b3, float* out) {
  int row = blockIdx.x * blockDim.x + threadIdx.x;
  if (row >= 8192) return;
  const __bf16* z = zH + (size_t)row * 192;
  float t1[96];
#pragma unroll
  for (int j = 0; j < 96; ++j) t1[j] = b1[j];
  for (int k = 0; k < 192; ++k) {
    float zv = (float)z[k];
#pragma unroll
    for (int j = 0; j < 96; ++j) t1[j] += zv * w1[k * 96 + j];
  }
  float t2[48];
#pragma unroll
  for (int j = 0; j < 48; ++j) t2[j] = b2[j];
  for (int k = 0; k < 96; ++k) {
    float v = fmaxf(t1[k], 0.0f);
#pragma unroll
    for (int j = 0; j < 48; ++j) t2[j] += v * w2[k * 48 + j];
  }
  float v = b3[0];
#pragma unroll
  for (int k = 0; k < 48; ++k) v += fmaxf(t2[k], 0.0f) * w3[k];
  out[row] = tanhf(v);
}

// ---------------------------------------------------------------------------
// Host launcher
// ---------------------------------------------------------------------------
extern "C" void kernel_launch(void* const* d_in, const int* in_sizes, int n_in,
                              void* d_out, int out_size, void* d_ws, size_t ws_size,
                              hipStream_t stream) {
  const float* x      = (const float*)d_in[0];
  const float* conv_w = (const float*)d_in[1];
  const float* conv_b = (const float*)d_in[2];
  const float* bn_s   = (const float*)d_in[3];
  const float* bn_b   = (const float*)d_in[4];
  const float* bn_m   = (const float*)d_in[5];
  const float* bn_v   = (const float*)d_in[6];
  const float* ep_w1  = (const float*)d_in[7];
  const float* ep_b1  = (const float*)d_in[8];
  const float* ep_w2  = (const float*)d_in[9];
  const float* ep_b2  = (const float*)d_in[10];
  const float* fc_w1  = (const float*)d_in[11];
  const float* fc_b1  = (const float*)d_in[12];
  const float* fc_w2  = (const float*)d_in[13];
  const float* fc_b2  = (const float*)d_in[14];
  const float* be_w1  = (const float*)d_in[15];
  const float* be_b1  = (const float*)d_in[16];
  const float* be_w2  = (const float*)d_in[17];
  const float* be_b2  = (const float*)d_in[18];
  const float* l_w1   = (const float*)d_in[19];
  const float* l_b1   = (const float*)d_in[20];
  const float* l_w2   = (const float*)d_in[21];
  const float* l_b2   = (const float*)d_in[22];
  const float* h_w1   = (const float*)d_in[23];
  const float* h_b1   = (const float*)d_in[24];
  const float* h_w2   = (const float*)d_in[25];
  const float* h_b2   = (const float*)d_in[26];
  const float* pol_w  = (const float*)d_in[27];
  const float* pol_b  = (const float*)d_in[28];
  const float* v_w1   = (const float*)d_in[29];
  const float* v_b1   = (const float*)d_in[30];
  const float* v_w2   = (const float*)d_in[31];
  const float* v_b2   = (const float*)d_in[32];
  const float* v_w3   = (const float*)d_in[33];
  const float* v_b3   = (const float*)d_in[34];

  char* wsp = (char*)d_ws;
  auto alloc = [&](size_t bytes) -> char* {
    char* p = wsp;
    wsp += (bytes + 255) & ~(size_t)255;
    return p;
  };
  __bf16* Mt    = (__bf16*)alloc((size_t)12288 * 64 * 2);
  float*  effb  = (float*)alloc(192 * 4);
  __bf16* fc1t  = (__bf16*)alloc((size_t)192 * 12352 * 2);
  __bf16* fc2t  = (__bf16*)alloc((size_t)192 * 192 * 2);
  __bf16* be1t  = (__bf16*)alloc((size_t)192 * 192 * 2);
  __bf16* be2t  = (__bf16*)alloc((size_t)192 * 192 * 2);
  __bf16* lW1t  = (__bf16*)alloc((size_t)192 * 384 * 2);
  __bf16* lw1ct = (__bf16*)alloc((size_t)192 * 192 * 2);
  __bf16* lW2t  = (__bf16*)alloc((size_t)192 * 192 * 2);
  __bf16* hW1t  = (__bf16*)alloc((size_t)192 * 384 * 2);
  __bf16* hW2t  = (__bf16*)alloc((size_t)192 * 192 * 2);
  __bf16* polt  = (__bf16*)alloc((size_t)4096 * 192 * 2);
  float*  lfbuf = (float*)alloc((size_t)8192 * 192 * 4);
  __bf16* zHbuf = (__bf16*)alloc((size_t)8192 * 192 * 2);
  if ((size_t)(wsp - (char*)d_ws) > ws_size) return;  // insufficient scratch

  k_conv_fold<<<(12288 * 64 + 255) / 256, 256, 0, stream>>>(Mt, effb, conv_w, conv_b, bn_s,
                                                            bn_b, bn_m, bn_v);
  auto T = [&](__bf16* dst, const float* src, int In, int Out, int Kpad) {
    long total = (long)Out * Kpad;
    k_transpose_bf16<<<(unsigned)((total + 255) / 256), 256, 0, stream>>>(dst, src, In, Out, Kpad);
  };
  T(fc1t, fc_w1, 12336, 192, 12352);
  T(fc2t, fc_w2, 192, 192, 192);
  T(be1t, be_w1, 192, 192, 192);
  T(be2t, be_w2, 192, 192, 192);
  T(lW1t, l_w1, 384, 192, 384);
  T(lw1ct, l_w1 + (size_t)384 * 192, 192, 192, 192);
  T(lW2t, l_w2, 192, 192, 192);
  T(hW1t, h_w1, 384, 192, 384);
  T(hW2t, h_w2, 192, 192, 192);
  T(polt, pol_w, 192, 4096, 192);

  k_encode<<<128, 128, 4 * 11776, stream>>>(x, ep_w1, ep_b1, ep_w2, ep_b2, Mt, effb, fc1t,
                                            fc_b1, fc2t, fc_b2, be1t, be_b1, be2t, be_b2,
                                            lw1ct, l_b1, lfbuf);
  k_recur<<<256, 64, 2 * 31488, stream>>>(lfbuf, lW1t, lW2t, l_b2, hW1t, hW2t, h_b1, h_b2,
                                          zHbuf);
  k_policy<<<64, 256, 0, stream>>>(zHbuf, polt, pol_b, (float*)d_out);
  k_value<<<32, 256, 0, stream>>>(zHbuf, v_w1, v_b1, v_w2, v_b2, v_w3, v_b3,
                                  (float*)d_out + (size_t)8192 * 4096);
}
